// MatryoshkaAttention_24300924960800
// MI455X (gfx1250) — compile-verified, matrix operand order checked by reading, not run on hardware
//
#include <hip/hip_runtime.h>
#include <cstdint>
#include <cstddef>

// ---------------------------------------------------------------------------
// MI455X (gfx1250) implementation. All heavy math via v_wmma_f32_16x16x32_bf16.
// Weights are kept transposed (NxK) so every WMMA fragment load is two
// contiguous 16-byte runs -> ds_load_b128 / global_load_b128, no scalar gathers.
// ---------------------------------------------------------------------------

typedef __bf16 bf16_t;
typedef __attribute__((ext_vector_type(16))) __bf16 v16bf;
typedef __attribute__((ext_vector_type(8)))  float  v8f;

#define RANK 8

static constexpr int Bdim = 4;
static constexpr int Tdim = 1024;
static constexpr int Ddim = 2048;
static constexpr int NQ   = 1536;      // total head dim across tiers
static constexpr int Hn   = 24;        // total heads (4 + 8 + 12), dk = 64 each
static constexpr int BT   = Bdim * Tdim;

// K index inside a 16-bit 16x32 WMMA A/B fragment for element slot e (0..15):
// per lane the 16 elements form two contiguous 8-element runs:
// [half*8, half*8+8) and [16+half*8, 16+half*8+8).
__device__ __forceinline__ int frag_k(int e, int half) {
    int v = e >> 1, sub = e & 1;
    return ((v < 4) ? 0 : 16) + half * 8 + (v & 3) * 2 + sub;
}

// ---------------------------------------------------------------------------
// Elementwise fp32 -> bf16 cast
// ---------------------------------------------------------------------------
__global__ void cast_bf16_kernel(const float* __restrict__ src,
                                 bf16_t* __restrict__ dst, size_t n) {
    size_t i = (size_t)blockIdx.x * blockDim.x + threadIdx.x;
    size_t stride = (size_t)gridDim.x * blockDim.x;
    for (; i < n; i += stride) dst[i] = (bf16_t)src[i];
}

// ---------------------------------------------------------------------------
// LDS-tiled cast + transpose: src fp32 (K x N, row-major) -> dst bf16 (N x K)
// ---------------------------------------------------------------------------
__global__ __launch_bounds__(256) void cast_transpose_kernel(
    const float* __restrict__ src, bf16_t* __restrict__ dst, int K, int N) {
    __shared__ float tile[32][33];
    const int k0 = blockIdx.y * 32;
    const int n0 = blockIdx.x * 32;
    const int tx = threadIdx.x;        // 0..31
    const int ty = threadIdx.y;        // 0..7
#pragma unroll
    for (int i = 0; i < 32; i += 8)
        tile[ty + i][tx] = src[(size_t)(k0 + ty + i) * N + n0 + tx];
    __syncthreads();
#pragma unroll
    for (int i = 0; i < 32; i += 8)
        dst[(size_t)(n0 + ty + i) * K + k0 + tx] = (bf16_t)tile[tx][ty + i];
}

// ---------------------------------------------------------------------------
// Tiled bf16 GEMM (NT): C(fp32, MxN) = A(bf16, MxK, row-major) x Bt(bf16, NxK)^T
// 128x128 block tile, 8 waves, each wave computes 32x64 via 2x4 WMMA tiles.
// Both LDS tiles are K-contiguous -> all fragment loads are ds_load_b128.
// ---------------------------------------------------------------------------
#define TILE_M 128
#define TILE_N 128
#define TILE_K 32

__global__ __launch_bounds__(256) void gemm_bf16_nt_kernel(
    const bf16_t* __restrict__ A, const bf16_t* __restrict__ Bt,
    float* __restrict__ C, int M, int N, int K) {
    __shared__ bf16_t As[TILE_M][TILE_K];   // 8 KB
    __shared__ bf16_t Bs[TILE_N][TILE_K];   // 8 KB (transposed: n-major)

    const int tid  = threadIdx.x;
    const int lane = tid & 31;
    const int wave = tid >> 5;       // 0..7
    const int wrow = wave & 3;       // 4 wave-rows of 32
    const int wcol = wave >> 2;      // 2 wave-cols of 64
    const int half = lane >> 4;
    const int l16  = lane & 15;

    const int m0 = blockIdx.y * TILE_M;
    const int n0 = blockIdx.x * TILE_N;

    v8f acc[2][4] = {};

    for (int k0 = 0; k0 < K; k0 += TILE_K) {
        // --- cooperative tile loads (16B per thread per step) ---
#pragma unroll
        for (int it = 0; it < 2; ++it) {
            int idx = tid + it * 256;            // 0..511 chunks of 8 bf16
            int r = idx >> 2;                    // 0..127
            int c = (idx & 3) * 8;               // 0,8,16,24
            *reinterpret_cast<uint4*>(&As[r][c]) =
                *reinterpret_cast<const uint4*>(&A[(size_t)(m0 + r) * K + k0 + c]);
            *reinterpret_cast<uint4*>(&Bs[r][c]) =
                *reinterpret_cast<const uint4*>(&Bt[(size_t)(n0 + r) * K + k0 + c]);
        }
        if (k0 + TILE_K < K) {       // hint the next tiles toward the caches
            __builtin_prefetch(&A[(size_t)(m0 + (tid >> 1)) * K + k0 + TILE_K]);
            __builtin_prefetch(&Bt[(size_t)(n0 + (tid >> 1)) * K + k0 + TILE_K]);
        }
        __syncthreads();

        // --- A fragments: 2 m-tiles of 16x32 ---
        v16bf afrag[2];
#pragma unroll
        for (int mt = 0; mt < 2; ++mt) {
            int am = wrow * 32 + mt * 16 + l16;
#pragma unroll
            for (int e = 0; e < 16; ++e) afrag[mt][e] = As[am][frag_k(e, half)];
        }
        // --- B fragments + WMMA: 4 n-tiles ---
#pragma unroll
        for (int nt = 0; nt < 4; ++nt) {
            int bn = wcol * 64 + nt * 16 + l16;
            v16bf bfrag;
#pragma unroll
            for (int e = 0; e < 16; ++e) bfrag[e] = Bs[bn][frag_k(e, half)];
#pragma unroll
            for (int mt = 0; mt < 2; ++mt) {
                acc[mt][nt] = __builtin_amdgcn_wmma_f32_16x16x32_bf16(
                    false, afrag[mt], false, bfrag, (short)0, acc[mt][nt], false, false);
            }
        }
        __syncthreads();
    }

    // --- store: C layout row m = r + 8*half, col n = lane&15 ---
#pragma unroll
    for (int mt = 0; mt < 2; ++mt)
#pragma unroll
        for (int nt = 0; nt < 4; ++nt)
#pragma unroll
            for (int r = 0; r < 8; ++r) {
                int m = m0 + wrow * 32 + mt * 16 + half * 8 + r;
                int n = n0 + wcol * 64 + nt * 16 + l16;
                C[(size_t)m * N + n] = acc[mt][nt][r];
            }
}

// ---------------------------------------------------------------------------
// Low-rank K/V correction: KV[bt, out_off + h*64 + d] += sum_r (x_hi . F[:,h,r]) * P[h,r,d]
// One block per (bt, head). Tiny FLOP count -> plain VALU kernel.
// ---------------------------------------------------------------------------
__global__ __launch_bounds__(256) void lowrank_correct_kernel(
    const float* __restrict__ x, const float* __restrict__ F,
    const float* __restrict__ P, float* __restrict__ KV,
    int in_off, int hi, int nh, int out_off) {
    __shared__ float red[256];
    __shared__ float c[RANK];
    const int bt = blockIdx.x;
    const int h  = blockIdx.y;
    const float* xr = x + (size_t)bt * Ddim + in_off;

    for (int r = 0; r < RANK; ++r) {
        float s = 0.f;
        for (int i = threadIdx.x; i < hi; i += 256)
            s += xr[i] * F[(size_t)i * (nh * RANK) + h * RANK + r];
        red[threadIdx.x] = s;
        __syncthreads();
        for (int k = 128; k > 0; k >>= 1) {
            if (threadIdx.x < k) red[threadIdx.x] += red[threadIdx.x + k];
            __syncthreads();
        }
        if (threadIdx.x == 0) c[r] = red[0];
        __syncthreads();
    }
    if (threadIdx.x < 64) {
        int d = threadIdx.x;
        float a = 0.f;
#pragma unroll
        for (int r = 0; r < RANK; ++r) a += c[r] * P[((size_t)h * RANK + r) * 64 + d];
        KV[(size_t)bt * NQ + out_off + h * 64 + d] += a;
    }
}

// ---------------------------------------------------------------------------
// Head-split + bf16 cast: [bt][1536] fp32 -> [b][24][t][64] bf16   (Q, K)
// ---------------------------------------------------------------------------
__global__ void split_heads_kernel(const float* __restrict__ src,
                                   bf16_t* __restrict__ dst) {
    size_t i = (size_t)blockIdx.x * blockDim.x + threadIdx.x;
    size_t n = (size_t)BT * NQ;
    size_t stride = (size_t)gridDim.x * blockDim.x;
    for (; i < n; i += stride) {
        int bt = (int)(i / NQ), col = (int)(i % NQ);
        int h = col >> 6, d = col & 63;
        int b = bt / Tdim, t = bt % Tdim;
        dst[((((size_t)b * Hn + h) * Tdim) + t) * 64 + d] = (bf16_t)src[i];
    }
}

// ---------------------------------------------------------------------------
// Head-split + bf16 + transpose: [bt][1536] fp32 -> [b][24][64][t] bf16  (V)
// Output-indexed so bf16 writes are fully coalesced; fp32 reads hit L2.
// ---------------------------------------------------------------------------
__global__ void split_heads_T_kernel(const float* __restrict__ src,
                                     bf16_t* __restrict__ dst) {
    size_t i = (size_t)blockIdx.x * blockDim.x + threadIdx.x;
    size_t n = (size_t)BT * NQ;
    size_t stride = (size_t)gridDim.x * blockDim.x;
    for (; i < n; i += stride) {
        int bh = (int)(i / (64 * Tdim));
        int rem = (int)(i % (64 * Tdim));
        int d = rem / Tdim, t = rem % Tdim;
        int b = bh / Hn, h = bh % Hn;
        dst[i] = (bf16_t)src[((size_t)b * Tdim + t) * NQ + h * 64 + d];
    }
}

// ---------------------------------------------------------------------------
// Causal flash attention, bf16 WMMA, dk = 64, one wave per 16-query tile.
// Q,K stored [b][h][t][64]; V stored transposed [b][h][64][t] so the P@V
// B-fragment is two contiguous b128 runs per lane.
// Output written directly in the concat layout: [bt][h*64 + d] (bf16).
// ---------------------------------------------------------------------------
__global__ __launch_bounds__(256) void flash_attn_kernel(
    const bf16_t* __restrict__ Qh, const bf16_t* __restrict__ Kh,
    const bf16_t* __restrict__ Vt, bf16_t* __restrict__ Oc) {
    __shared__ bf16_t Pst[8][16][32];   // per-wave P staging (8 KB)

    const int lane = threadIdx.x & 31;
    const int wave = threadIdx.x >> 5;
    const int half = lane >> 4;
    const int l16  = lane & 15;

    const int bh = blockIdx.x;             // b*24 + h
    const int b  = bh / Hn;
    const int h  = bh % Hn;
    const int q0 = (blockIdx.y * 8 + wave) * 16;

    const bf16_t* Qp = Qh + (size_t)bh * Tdim * 64;
    const bf16_t* Kp = Kh + (size_t)bh * Tdim * 64;
    const bf16_t* Vp = Vt + (size_t)bh * Tdim * 64;   // [64][Tdim]

    // Q fragments (16 queries x 64 d, two 32-wide K chunks)
    v16bf qf[2];
#pragma unroll
    for (int dc = 0; dc < 2; ++dc)
#pragma unroll
        for (int e = 0; e < 16; ++e)
            qf[dc][e] = Qp[(size_t)(q0 + l16) * 64 + dc * 32 + frag_k(e, half)];

    v8f o[4] = {};
    float m_i[8], l_i[8];
#pragma unroll
    for (int r = 0; r < 8; ++r) { m_i[r] = -1e30f; l_i[r] = 0.f; }

    const float scale = 0.125f;   // 1/sqrt(64)

    for (int kb = 0; kb < q0 + 16; kb += 32) {
        // ---- S = Q K^T for 16x32 key block (2 C tiles x 2 d chunks) ----
        v8f s[2] = {};
#pragma unroll
        for (int st = 0; st < 2; ++st) {
#pragma unroll
            for (int dc = 0; dc < 2; ++dc) {
                v16bf kf;
                int key = kb + st * 16 + l16;
#pragma unroll
                for (int e = 0; e < 16; ++e)
                    kf[e] = Kp[(size_t)key * 64 + dc * 32 + frag_k(e, half)];
                s[st] = __builtin_amdgcn_wmma_f32_16x16x32_bf16(
                    false, qf[dc], false, kf, (short)0, s[st], false, false);
            }
        }
        // ---- online softmax (rows m = r + 8*half, 16 lanes per row) ----
#pragma unroll
        for (int r = 0; r < 8; ++r) {
            int gq = q0 + r + 8 * half;
            float best = -1e30f;
#pragma unroll
            for (int st = 0; st < 2; ++st) {
                int key = kb + st * 16 + l16;
                float val = s[st][r] * scale;
                val = (key <= gq) ? val : -1e30f;
                s[st][r] = val;
                best = fmaxf(best, val);
            }
#pragma unroll
            for (int off = 1; off < 16; off <<= 1)
                best = fmaxf(best, __shfl_xor(best, off, 32));
            float m_new = fmaxf(m_i[r], best);
            float corr  = __expf(m_i[r] - m_new);
            float rowsum = 0.f;
#pragma unroll
            for (int st = 0; st < 2; ++st) {
                float p = __expf(s[st][r] - m_new);
                s[st][r] = p;
                rowsum += p;
            }
#pragma unroll
            for (int off = 1; off < 16; off <<= 1)
                rowsum += __shfl_xor(rowsum, off, 32);
            l_i[r] = l_i[r] * corr + rowsum;
            m_i[r] = m_new;
#pragma unroll
            for (int nt = 0; nt < 4; ++nt) o[nt][r] *= corr;
        }
        // ---- reshape P: C layout -> A fragment via LDS round-trip ----
#pragma unroll
        for (int st = 0; st < 2; ++st)
#pragma unroll
            for (int r = 0; r < 8; ++r)
                Pst[wave][r + 8 * half][st * 16 + l16] = (bf16_t)s[st][r];
        asm volatile("s_wait_dscnt 0" ::: "memory");
        v16bf pf;
#pragma unroll
        for (int e = 0; e < 16; ++e) pf[e] = Pst[wave][l16][frag_k(e, half)];
        // ---- O += P V (K-dim = 32 keys, 4 d-tiles of 16), V is d-major ----
#pragma unroll
        for (int nt = 0; nt < 4; ++nt) {
            v16bf vf;
#pragma unroll
            for (int e = 0; e < 16; ++e)
                vf[e] = Vp[(size_t)(nt * 16 + l16) * Tdim + kb + frag_k(e, half)];
            o[nt] = __builtin_amdgcn_wmma_f32_16x16x32_bf16(
                false, pf, false, vf, (short)0, o[nt], false, false);
        }
        asm volatile("s_wait_dscnt 0" ::: "memory");   // P reads done before overwrite
    }
    // ---- normalize + store in concat layout ----
#pragma unroll
    for (int nt = 0; nt < 4; ++nt)
#pragma unroll
        for (int r = 0; r < 8; ++r) {
            int q = q0 + r + 8 * half;
            int d = nt * 16 + l16;
            float val = o[nt][r] / l_i[r];
            Oc[((size_t)b * Tdim + q) * NQ + h * 64 + d] = (bf16_t)val;
        }
}

// ---------------------------------------------------------------------------
// Host-side orchestration
// ---------------------------------------------------------------------------
extern "C" void kernel_launch(void* const* d_in, const int* in_sizes, int n_in,
                              void* d_out, int out_size, void* d_ws, size_t ws_size,
                              hipStream_t stream) {
    (void)in_sizes; (void)n_in; (void)out_size; (void)ws_size;
    const float* x   = (const float*)d_in[0];
    const float* W_Q = (const float*)d_in[1];
    const float* W_K = (const float*)d_in[2];
    const float* W_V = (const float*)d_in[3];
    const float* W_O = (const float*)d_in[4];
    const float* FK0 = (const float*)d_in[5];
    const float* PK0 = (const float*)d_in[6];
    const float* FV0 = (const float*)d_in[7];
    const float* PV0 = (const float*)d_in[8];
    const float* FK1 = (const float*)d_in[9];
    const float* PK1 = (const float*)d_in[10];
    const float* FV1 = (const float*)d_in[11];
    const float* PV1 = (const float*)d_in[12];
    float* out = (float*)d_out;

    char* wsp = (char*)d_ws;
    auto alloc = [&](size_t bytes) {
        char* p = wsp; wsp += (bytes + 255) & ~(size_t)255; return p;
    };
    bf16_t* x16  = (bf16_t*)alloc((size_t)BT * Ddim * 2);
    bf16_t* wqT  = (bf16_t*)alloc((size_t)Ddim * NQ * 2);   // [NQ][D]
    bf16_t* wkT  = (bf16_t*)alloc((size_t)Ddim * NQ * 2);
    bf16_t* wvT  = (bf16_t*)alloc((size_t)Ddim * NQ * 2);
    bf16_t* woT  = (bf16_t*)alloc((size_t)NQ * Ddim * 2);   // [D][NQ]
    float*  Qf   = (float*)alloc((size_t)BT * NQ * 4);
    float*  Kf   = (float*)alloc((size_t)BT * NQ * 4);
    float*  Vf   = (float*)alloc((size_t)BT * NQ * 4);
    bf16_t* Qh   = (bf16_t*)alloc((size_t)BT * NQ * 2);
    bf16_t* Kh   = (bf16_t*)alloc((size_t)BT * NQ * 2);
    bf16_t* Vtr  = (bf16_t*)alloc((size_t)BT * NQ * 2);
    bf16_t* att  = (bf16_t*)alloc((size_t)BT * NQ * 2);

    // 1. fp32 -> bf16 casts (weights cast + transposed to NxK)
    cast_bf16_kernel<<<4096, 256, 0, stream>>>(x, x16, (size_t)BT * Ddim);
    {
        dim3 blk(32, 8);
        dim3 gW(NQ / 32, Ddim / 32);      // W_Q/K/V: (2048 x 1536) -> (1536 x 2048)
        cast_transpose_kernel<<<gW, blk, 0, stream>>>(W_Q, wqT, Ddim, NQ);
        cast_transpose_kernel<<<gW, blk, 0, stream>>>(W_K, wkT, Ddim, NQ);
        cast_transpose_kernel<<<gW, blk, 0, stream>>>(W_V, wvT, Ddim, NQ);
        dim3 gO(Ddim / 32, NQ / 32);      // W_O: (1536 x 2048) -> (2048 x 1536)
        cast_transpose_kernel<<<gO, blk, 0, stream>>>(W_O, woT, NQ, Ddim);
    }

    // 2. QKV projections (WMMA, NT)
    dim3 gQKV(NQ / TILE_N, BT / TILE_M);
    gemm_bf16_nt_kernel<<<gQKV, 256, 0, stream>>>(x16, wqT, Qf, BT, NQ, Ddim);
    gemm_bf16_nt_kernel<<<gQKV, 256, 0, stream>>>(x16, wkT, Kf, BT, NQ, Ddim);
    gemm_bf16_nt_kernel<<<gQKV, 256, 0, stream>>>(x16, wvT, Vf, BT, NQ, Ddim);

    // 3. low-rank K/V corrections (tier0: hi=1792 from col 256; tier1: hi=1024 from col 1024)
    lowrank_correct_kernel<<<dim3(BT, 4), 256, 0, stream>>>(x, FK0, PK0, Kf, 256, 1792, 4, 0);
    lowrank_correct_kernel<<<dim3(BT, 4), 256, 0, stream>>>(x, FV0, PV0, Vf, 256, 1792, 4, 0);
    lowrank_correct_kernel<<<dim3(BT, 8), 256, 0, stream>>>(x, FK1, PK1, Kf, 1024, 1024, 8, 256);
    lowrank_correct_kernel<<<dim3(BT, 8), 256, 0, stream>>>(x, FV1, PV1, Vf, 1024, 1024, 8, 256);

    // 4. head-split + bf16 (Q,K row-major; V d-major for transposed fragments)
    split_heads_kernel<<<8192, 256, 0, stream>>>(Qf, Qh);
    split_heads_kernel<<<8192, 256, 0, stream>>>(Kf, Kh);
    split_heads_T_kernel<<<8192, 256, 0, stream>>>(Vf, Vtr);

    // 5. causal flash attention (WMMA): 96 (b,h) pairs x 8 blocks of 8 q-tiles
    flash_attn_kernel<<<dim3(Bdim * Hn, Tdim / (16 * 8)), 256, 0, stream>>>(Qh, Kh, Vtr, att);

    // 6. output projection (WMMA, NT) -> fp32 d_out
    dim3 gOut(Ddim / TILE_N, BT / TILE_M);
    gemm_bf16_nt_kernel<<<gOut, 256, 0, stream>>>(att, woT, out, BT, Ddim, NQ);
}
